// ScaledDotProductAttention_16398185136643
// MI455X (gfx1250) — matrix operand order repaired and checked by reading, not compiled, not run on hardware
//
#include <hip/hip_runtime.h>
#include <hip/hip_bf16.h>
#include <math.h>

typedef __attribute__((ext_vector_type(16))) __bf16 v16bf;
typedef __attribute__((ext_vector_type(8)))  float  v8f;
typedef __attribute__((ext_vector_type(4)))  float  f4;

#define NB   16
#define TQ   2048
#define TK   512
#define DD   128
#define SROW 516                      // 512 + 4 pad: bank-conflict-free frag reads
#define INV_T 0.08838834764831845f    // 1 / 11.313708498984761

__global__ __launch_bounds__(32)
void attn_fused_kernel(const float* __restrict__ q,
                       const float* __restrict__ k,
                       const float* __restrict__ v,
                       const float* __restrict__ key_mask,
                       const float* __restrict__ query_mask,
                       const float* __restrict__ mapping_mask,
                       const float* __restrict__ attn_prior,
                       float* __restrict__ out,
                       float* __restrict__ attn_o,
                       float* __restrict__ attn_raw_o,
                       float* __restrict__ logprob_o)
{
    __shared__ float S[16][SROW];     // 33 KB score/probability tile

    const int L  = threadIdx.x;       // lane 0..31 (wave32)
    const int m  = L & 15;            // row (A/C) or column (B/C) index
    const int hh = L >> 4;            // half-wave select
    const int b  = blockIdx.x >> 7;   // batch
    const int m0 = (blockIdx.x & 127) * 16;  // first q row of this tile

    // ---- Q A-fragments (16x32 bf16 layout), loaded once -------------------
    const float* qrow = q + ((size_t)b * TQ + m0 + m) * DD;
    v16bf aq[4];
    #pragma unroll
    for (int kc = 0; kc < 4; ++kc) {
        const int c0 = kc * 32 + hh * 8;
        #pragma unroll
        for (int j = 0; j < 8; ++j) {
            aq[kc][j]     = (__bf16)qrow[c0 + j];
            aq[kc][j + 8] = (__bf16)qrow[c0 + 16 + j];
        }
    }

    // ---- Phase A: S = keymask(Q K^T / T) into LDS -------------------------
    for (int nt = 0; nt < 32; ++nt) {
        const int k0 = nt * 16;
        const float* krow = k + ((size_t)b * TK + k0 + m) * DD;
        const float  km   = key_mask[(size_t)b * TK + k0 + m];
        v8f c = {0.f, 0.f, 0.f, 0.f, 0.f, 0.f, 0.f, 0.f};
        #pragma unroll
        for (int kc = 0; kc < 4; ++kc) {
            v16bf bk;
            const int c0 = kc * 32 + hh * 8;    // B: K c0..c0+7 and c0+16..c0+23
            #pragma unroll
            for (int j = 0; j < 16; ++j) bk[j] = (__bf16)krow[c0 + (j < 8 ? j : j + 8)];
            c = __builtin_amdgcn_wmma_f32_16x16x32_bf16(
                    false, aq[kc], false, bk, (short)0, c, false, false);
        }
        // C element column == k0+m (this lane's own key) -> its mask is km
        #pragma unroll
        for (int r = 0; r < 8; ++r) {
            const float s = (km == 0.f) ? -INFINITY : c[r] * INV_T;
            S[r + hh * 8][k0 + m] = s;
        }
    }
    __syncthreads();

    // ---- Phase B1: per-row logsumexp of S (lane owns row m, half hh) ------
    float Ls;
    {
        const float* row = &S[m][hh * 256];
        float mx = -INFINITY;
        for (int i = 0; i < 256; i += 4) {
            const f4 x = *(const f4*)(row + i);
            mx = fmaxf(mx, fmaxf(fmaxf(x.x, x.y), fmaxf(x.z, x.w)));
        }
        mx = fmaxf(mx, __shfl_xor(mx, 16, 32));
        float z = 0.f;
        for (int i = 0; i < 256; i += 4) {
            const f4 x = *(const f4*)(row + i);
            z += __expf(x.x - mx) + __expf(x.y - mx)
               + __expf(x.z - mx) + __expf(x.w - mx);
        }
        z += __shfl_xor(z, 16, 32);
        Ls = mx + __logf(z);
    }
    __syncthreads();

    // ---- Phase A2: U = S + log(prior + 1e-8); prior is [b][k][q] ----------
    {
        const float* pb = attn_prior + (size_t)b * TK * TQ + m0;
        for (int kr = 0; kr < TK; kr += 2) {
            const float pv =
                __builtin_nontemporal_load(pb + (size_t)(kr + hh) * TQ + m);
            S[m][kr + hh] += __logf(pv + 1e-8f);
        }
    }
    __syncthreads();

    // ---- Phase B2: per-row max/sumexp of U (softmax stats) ----------------
    float Mu, iZu;
    {
        const float* row = &S[m][hh * 256];
        float mx = -INFINITY;
        for (int i = 0; i < 256; i += 4) {
            const f4 x = *(const f4*)(row + i);
            mx = fmaxf(mx, fmaxf(fmaxf(x.x, x.y), fmaxf(x.z, x.w)));
        }
        mx = fmaxf(mx, __shfl_xor(mx, 16, 32));
        float z = 0.f;
        for (int i = 0; i < 256; i += 4) {
            const f4 x = *(const f4*)(row + i);
            z += __expf(x.x - mx) + __expf(x.y - mx)
               + __expf(x.z - mx) + __expf(x.w - mx);
        }
        z += __shfl_xor(z, 16, 32);
        Mu  = mx;
        iZu = 1.f / z;
    }
    __syncthreads();

    // ---- Phase C: stream outputs. float4 per lane -> b128 transactions,
    //      non-temporal: pure streaming data, keep L2 for q/k/v reuse. ------
    for (int mr = 0; mr < 16; ++mr) {
        const float Lsr = __shfl(Ls,  mr, 32);
        const float Mur = __shfl(Mu,  mr, 32);
        const float iZr = __shfl(iZu, mr, 32);
        const size_t rowg = (size_t)b * TQ + m0 + mr;
        const float  qm   = query_mask[rowg];
        const float* mmr  = mapping_mask + rowg * TK;
        const size_t o3   = rowg * TK;
        #pragma unroll
        for (int kc = 0; kc < 4; ++kc) {
            const int kcol = kc * 128 + L * 4;
            const f4 u4  = *(const f4*)(&S[mr][kcol]);
            const f4 mm4 = __builtin_nontemporal_load((const f4*)(mmr + kcol));
            f4 w4, pr4, pm4;
            w4.x = u4.x - Lsr;  w4.y = u4.y - Lsr;
            w4.z = u4.z - Lsr;  w4.w = u4.w - Lsr;
            pr4.x = __expf(u4.x - Mur) * iZr * qm;
            pr4.y = __expf(u4.y - Mur) * iZr * qm;
            pr4.z = __expf(u4.z - Mur) * iZr * qm;
            pr4.w = __expf(u4.w - Mur) * iZr * qm;
            pm4.x = pr4.x * mm4.x;  pm4.y = pr4.y * mm4.y;
            pm4.z = pr4.z * mm4.z;  pm4.w = pr4.w * mm4.w;
            __builtin_nontemporal_store(w4,  (f4*)(logprob_o  + o3 + kcol));
            __builtin_nontemporal_store(pr4, (f4*)(attn_raw_o + o3 + kcol));
            __builtin_nontemporal_store(pm4, (f4*)(attn_o     + o3 + kcol));
            *(f4*)(&S[mr][kcol]) = pm4;        // P for the PV matmul
        }
    }
    __syncthreads();

    // ---- Phase D: out = P V via bf16 WMMA ---------------------------------
    v8f acc[8];
    #pragma unroll
    for (int ct = 0; ct < 8; ++ct)
        acc[ct] = (v8f){0.f, 0.f, 0.f, 0.f, 0.f, 0.f, 0.f, 0.f};

    for (int kc = 0; kc < 16; ++kc) {
        v16bf ap;                                     // A frag from LDS P
        const float* prow = &S[m][kc * 32 + hh * 8];
        #pragma unroll
        for (int j = 0; j < 8; ++j) {
            ap[j]     = (__bf16)prow[j];
            ap[j + 8] = (__bf16)prow[16 + j];
        }
        const float* vb = v + ((size_t)b * TK + kc * 32 + hh * 8) * DD + m;
        #pragma unroll
        for (int ct = 0; ct < 8; ++ct) {
            v16bf bv;                                 // B frag: V columns
            const float* vp = vb + ct * 16;
            #pragma unroll
            for (int j = 0; j < 16; ++j) bv[j] = (__bf16)vp[(size_t)(j < 8 ? j : j + 8) * DD];
            acc[ct] = __builtin_amdgcn_wmma_f32_16x16x32_bf16(
                          false, ap, false, bv, (short)0, acc[ct], false, false);
        }
    }

    // ---- Phase E: store output tile (lanes 0..15 contiguous in d) ---------
    #pragma unroll
    for (int ct = 0; ct < 8; ++ct)
        #pragma unroll
        for (int r = 0; r < 8; ++r)
            out[((size_t)b * TQ + m0 + r + hh * 8) * DD + ct * 16 + m] = acc[ct][r];
}

extern "C" void kernel_launch(void* const* d_in, const int* in_sizes, int n_in,
                              void* d_out, int out_size, void* d_ws, size_t ws_size,
                              hipStream_t stream) {
    const float* q            = (const float*)d_in[0];
    const float* k            = (const float*)d_in[1];
    const float* v            = (const float*)d_in[2];
    const float* key_mask     = (const float*)d_in[3];
    const float* query_mask   = (const float*)d_in[4];
    const float* mapping_mask = (const float*)d_in[5];
    const float* attn_prior   = (const float*)d_in[6];

    float* out        = (float*)d_out;                          // B*TQ*D
    float* attn_o     = out      + (size_t)NB * TQ * DD;        // B*TQ*TK
    float* attn_raw_o = attn_o   + (size_t)NB * TQ * TK;        // B*TQ*TK
    float* logprob_o  = attn_raw_o + (size_t)NB * TQ * TK;      // B*1*TQ*TK

    dim3 grid(NB * (TQ / 16));   // 2048 waves, one 16-row q-tile each
    dim3 block(32);              // one wave32 per workgroup (33 KB LDS each)
    attn_fused_kernel<<<grid, block, 0, stream>>>(
        q, k, v, key_mask, query_mask, mapping_mask, attn_prior,
        out, attn_o, attn_raw_o, logprob_o);
}
